// nnmodel_83056077570906
// MI455X (gfx1250) — matrix operand order repaired and checked by reading, not compiled, not run on hardware
//
#include <hip/hip_runtime.h>

typedef __attribute__((ext_vector_type(16))) _Float16 v16h;
typedef __attribute__((ext_vector_type(8)))  _Float16 v8h;
typedef __attribute__((ext_vector_type(8)))  float    v8f;
typedef __attribute__((ext_vector_type(4)))  float    v4f;

#define N_IN   40
#define N_HID  80
#define N_OUT  40
#define BATCH  1048576
#define WAVES  4
#define BLOCK  (WAVES * 32)
#define ROWS_PER_WAVE 16
#define ROWS_PER_BLOCK (WAVES * ROWS_PER_WAVE)   // 64
#define NUM_TILES (BATCH / ROWS_PER_BLOCK)       // 16384
#define GRID   2048
#define YSTRIDE 84                                // fp32 dwords; 84 mod 64 spreads banks

// d_ws layout (element offsets):
//   _Float16 w1h[80*40], w1l[80*40]   : Wt1[n][k] = W_enc[k][n] split hi/lo
//   _Float16 w2h[48*80], w2l[48*80]   : Wt2[n][k] = W_comb[k][n] split hi/lo (n>=40 zero)
//   float    b1[80], b2[48]

__device__ inline v8f wmma16(v16h a, v16h b, v8f c) {
  return __builtin_amdgcn_wmma_f32_16x16x32_f16(false, a, false, b, (short)0, c, false, false);
}

__device__ inline v8f splat8(float b) {
  v8f r;
#pragma unroll
  for (int i = 0; i < 8; ++i) r[i] = b;
  return r;
}

// 16 contiguous f16 (two aligned 16B LDS loads) -> v16h
__device__ inline v16h ld16(const _Float16* p) {
  v8h a = *(const v8h*)p;
  v8h b = *(const v8h*)(p + 8);
  v16h r;
#pragma unroll
  for (int i = 0; i < 8; ++i) { r[i] = a[i]; r[i + 8] = b[i]; }
  return r;
}

// ---------------- setup: fold graph conv + decoder into W_comb -------------
__global__ void setup_kernel(const float* __restrict__ W_enc, const float* __restrict__ b_enc,
                             const float* __restrict__ W_rel, const float* __restrict__ b_rel,
                             const float* __restrict__ W_root, const float* __restrict__ W_dec,
                             const float* __restrict__ b_dec, void* ws) {
  _Float16* w1h = (_Float16*)ws;
  _Float16* w1l = w1h + 80 * 40;
  _Float16* w2h = w1l + 80 * 40;
  _Float16* w2l = w2h + 48 * 80;
  float*    b1  = (float*)(w2l + 48 * 80);
  float*    b2  = b1 + 80;
  const float aw = expf(-1.0f / 9.0f);

  for (int i = threadIdx.x; i < 80 * 40; i += blockDim.x) {   // Wt1[n][k]
    int n = i / 40, k = i % 40;
    float v = W_enc[k * 80 + n];
    _Float16 h = (_Float16)v;
    w1h[i] = h;
    w1l[i] = (_Float16)(v - (float)h);
  }
  for (int i = threadIdx.x; i < 48 * 80; i += blockDim.x) {   // Wt2[n][k]
    int n = i / 80, k = i % 80;
    float v = 0.0f;
    if (n < 40) {
      int s = k >> 3, f = k & 7;
      int sp = (s + 1) % 10, sm = (s + 9) % 10;
      for (int g = 0; g < 8; ++g) {
        v += aw * W_rel[f * 8 + g] *
             (W_dec[(sp * 8 + g) * 40 + n] + W_dec[(sm * 8 + g) * 40 + n]);
        v += W_root[f * 8 + g] * W_dec[(s * 8 + g) * 40 + n];
      }
    }
    _Float16 h = (_Float16)v;
    w2h[i] = h;
    w2l[i] = (_Float16)(v - (float)h);
  }
  for (int n = threadIdx.x; n < 80; n += blockDim.x) b1[n] = b_enc[n];
  for (int n = threadIdx.x; n < 48; n += blockDim.x) {
    float v = 0.0f;
    if (n < 40) {
      v = b_dec[n];
      for (int t = 0; t < 10; ++t)
        for (int g = 0; g < 8; ++g)
          v += b_rel[g] * W_dec[(t * 8 + g) * 40 + n];
    }
    b2[n] = v;
  }
}

// ---------------- fused MLP: relu(relu(x@W1+b1)@W2+b2) --------------------
__global__ __launch_bounds__(BLOCK) void fused_mlp_kernel(const float* __restrict__ x,
                                                          const void* __restrict__ ws,
                                                          float* __restrict__ out) {
  __shared__ _Float16 sW1h[80 * 40];
  __shared__ _Float16 sW1l[80 * 40];
  __shared__ _Float16 sW2h[48 * 80];
  __shared__ _Float16 sW2l[48 * 80];
  __shared__ float    sB1[80];
  __shared__ float    sB2[48];
  __shared__ float    sY[WAVES][16 * YSTRIDE];

  {  // stage weights global -> LDS (dword copies)
    const uint32_t* g1h = (const uint32_t*)ws;
    const uint32_t* g1l = g1h + 1600;
    const uint32_t* g2h = g1l + 1600;
    const uint32_t* g2l = g2h + 1920;
    const float*    gb1 = (const float*)(g2l + 1920);
    const float*    gb2 = gb1 + 80;
    uint32_t* l1h = (uint32_t*)sW1h;
    uint32_t* l1l = (uint32_t*)sW1l;
    uint32_t* l2h = (uint32_t*)sW2h;
    uint32_t* l2l = (uint32_t*)sW2l;
    for (int i = threadIdx.x; i < 1600; i += BLOCK) { l1h[i] = g1h[i]; l1l[i] = g1l[i]; }
    for (int i = threadIdx.x; i < 1920; i += BLOCK) { l2h[i] = g2h[i]; l2l[i] = g2l[i]; }
    for (int i = threadIdx.x; i < 80; i += BLOCK) sB1[i] = gb1[i];
    for (int i = threadIdx.x; i < 48; i += BLOCK) sB2[i] = gb2[i];
  }
  __syncthreads();

  const int lane = threadIdx.x & 31;
  const int wave = threadIdx.x >> 5;
  const int m    = lane & 15;   // row (A/C) or column (B/C) within tile
  const int half = lane >> 4;   // lane-half per ISA fragment layout
  float* ybuf = &sY[wave][0];

  for (int tile = blockIdx.x; tile < NUM_TILES; tile += GRID) {
    const int row0 = tile * ROWS_PER_BLOCK + wave * ROWS_PER_WAVE;
    const float* xr = x + (size_t)(row0 + m) * N_IN;

    {  // prefetch next tile's rows
      int nt = tile + GRID;
      if (nt < NUM_TILES)
        __builtin_prefetch(x + (size_t)(nt * ROWS_PER_BLOCK + wave * 16 + m) * N_IN + half * 20, 0, 0);
    }

    // ---- A fragments for GEMM1 (x, K=40 padded to 64), hi/lo split ----
    v16h a1h[2], a1l[2];
    {
      const int ka = half ? 16 : 0, kb = ka + 8;
      v4f p0 = *(const v4f*)(xr + ka);
      v4f p1 = *(const v4f*)(xr + ka + 4);
      v4f p2 = *(const v4f*)(xr + kb);
      v4f p3 = *(const v4f*)(xr + kb + 4);
      v16h h = {}, l = {};
#pragma unroll
      for (int i = 0; i < 4; ++i) {
        _Float16 t;
        t = (_Float16)p0[i]; h[i]      = t; l[i]      = (_Float16)(p0[i] - (float)t);
        t = (_Float16)p1[i]; h[4 + i]  = t; l[4 + i]  = (_Float16)(p1[i] - (float)t);
        t = (_Float16)p2[i]; h[8 + i]  = t; l[8 + i]  = (_Float16)(p2[i] - (float)t);
        t = (_Float16)p3[i]; h[12 + i] = t; l[12 + i] = (_Float16)(p3[i] - (float)t);
      }
      a1h[0] = h; a1l[0] = l;
    }
    {  // chunk K=32..63: only K=32..39 (half 0) nonzero
      v16h h = {}, l = {};
      if (!half) {
        v4f p0 = *(const v4f*)(xr + 32);
        v4f p1 = *(const v4f*)(xr + 36);
#pragma unroll
        for (int i = 0; i < 4; ++i) {
          _Float16 t;
          t = (_Float16)p0[i]; h[i]     = t; l[i]     = (_Float16)(p0[i] - (float)t);
          t = (_Float16)p1[i]; h[4 + i] = t; l[4 + i] = (_Float16)(p1[i] - (float)t);
        }
      }
      a1h[1] = h; a1l[1] = l;
    }

    // ---- GEMM1: 5 N-tiles x 2 K-chunks x 3 split terms = 30 WMMAs ----
    v8f acc1[5];
#pragma unroll
    for (int nt = 0; nt < 5; ++nt) acc1[nt] = splat8(sB1[nt * 16 + m]);
#pragma unroll
    for (int kc = 0; kc < 2; ++kc) {
#pragma unroll
      for (int nt = 0; nt < 5; ++nt) {
        const int n = nt * 16 + m;
        v16h bh = {}, bl = {};
        if (kc == 0) {
          bh = ld16(&sW1h[n * 40 + half * 16]);
          bl = ld16(&sW1l[n * 40 + half * 16]);
        } else if (!half) {  // B lanes 0-15 hold K=32..47; only 32..39 valid
          v8h t = *(const v8h*)&sW1h[n * 40 + 32];
          v8h u = *(const v8h*)&sW1l[n * 40 + 32];
#pragma unroll
          for (int i = 0; i < 8; ++i) { bh[i] = t[i]; bl[i] = u[i]; }
        }
        acc1[nt] = wmma16(a1h[kc], bh, acc1[nt]);
        acc1[nt] = wmma16(a1h[kc], bl, acc1[nt]);
        acc1[nt] = wmma16(a1l[kc], bh, acc1[nt]);
      }
    }

    // ---- ReLU + cross-lane transpose via per-wave LDS scratch ----
#pragma unroll
    for (int nt = 0; nt < 5; ++nt) {
      const int n = nt * 16 + m;
#pragma unroll
      for (int r = 0; r < 8; ++r) {
        float v = acc1[nt][r];
        ybuf[(r + half * 8) * YSTRIDE + n] = v > 0.0f ? v : 0.0f;
      }
    }
    asm volatile("s_wait_dscnt 0" ::: "memory");  // per-wave LDS RAW ordering

    // ---- GEMM2: y[16x80] @ W2[80x40pad48]; 3x3x3 = 27 WMMAs ----
    v8f acc2[3];
#pragma unroll
    for (int nt = 0; nt < 3; ++nt) acc2[nt] = splat8(sB2[nt * 16 + m]);

    const float* yr = ybuf + m * YSTRIDE;
#pragma unroll
    for (int kc = 0; kc < 3; ++kc) {
      v16h ah = {}, al = {};
      if (kc < 2) {
        const int ka = kc * 32 + half * 16, kb = ka + 8;
        v4f p0 = *(const v4f*)(yr + ka);
        v4f p1 = *(const v4f*)(yr + ka + 4);
        v4f p2 = *(const v4f*)(yr + kb);
        v4f p3 = *(const v4f*)(yr + kb + 4);
#pragma unroll
        for (int i = 0; i < 4; ++i) {
          _Float16 t;
          t = (_Float16)p0[i]; ah[i]      = t; al[i]      = (_Float16)(p0[i] - (float)t);
          t = (_Float16)p1[i]; ah[4 + i]  = t; al[4 + i]  = (_Float16)(p1[i] - (float)t);
          t = (_Float16)p2[i]; ah[8 + i]  = t; al[8 + i]  = (_Float16)(p2[i] - (float)t);
          t = (_Float16)p3[i]; ah[12 + i] = t; al[12 + i] = (_Float16)(p3[i] - (float)t);
        }
      } else {  // A: K=64..71 (half0) / 72..79 (half1) valid; upper 8 zero
        const int ka = 64 + half * 8;
        v4f p0 = *(const v4f*)(yr + ka);
        v4f p1 = *(const v4f*)(yr + ka + 4);
#pragma unroll
        for (int i = 0; i < 4; ++i) {
          _Float16 t;
          t = (_Float16)p0[i]; ah[i]     = t; al[i]     = (_Float16)(p0[i] - (float)t);
          t = (_Float16)p1[i]; ah[4 + i] = t; al[4 + i] = (_Float16)(p1[i] - (float)t);
        }
      }
#pragma unroll
      for (int nt = 0; nt < 3; ++nt) {
        const int n = nt * 16 + m;
        v16h bh = {}, bl = {};
        if (kc < 2) {
          const int kb0 = kc * 32 + half * 16;
          bh = ld16(&sW2h[n * 80 + kb0]);
          bl = ld16(&sW2l[n * 80 + kb0]);
        } else {  // B: K=64..71 (half0) / 72..79 (half1) valid; upper 8 zero
          v8h t = *(const v8h*)&sW2h[n * 80 + 64 + half * 8];
          v8h u = *(const v8h*)&sW2l[n * 80 + 64 + half * 8];
#pragma unroll
          for (int i = 0; i < 8; ++i) { bh[i] = t[i]; bl[i] = u[i]; }
        }
        acc2[nt] = wmma16(ah, bh, acc2[nt]);
        acc2[nt] = wmma16(ah, bl, acc2[nt]);
        acc2[nt] = wmma16(al, bh, acc2[nt]);
      }
    }

    // ---- ReLU + store out[B,40]; lanes 0-15 give 64B-coalesced runs ----
#pragma unroll
    for (int nt = 0; nt < 3; ++nt) {
      const int n = nt * 16 + m;
      if (n < N_OUT) {
#pragma unroll
        for (int r = 0; r < 8; ++r) {
          float v = acc2[nt][r];
          out[(size_t)(row0 + r + half * 8) * N_OUT + n] = v > 0.0f ? v : 0.0f;
        }
      }
    }
  }
}

extern "C" void kernel_launch(void* const* d_in, const int* in_sizes, int n_in,
                              void* d_out, int out_size, void* d_ws, size_t ws_size,
                              hipStream_t stream) {
  const float* x      = (const float*)d_in[0];
  const float* W_enc  = (const float*)d_in[1];
  const float* b_enc  = (const float*)d_in[2];
  const float* W_rel  = (const float*)d_in[3];
  const float* b_rel  = (const float*)d_in[4];
  const float* W_root = (const float*)d_in[5];
  const float* W_dec  = (const float*)d_in[6];
  const float* b_dec  = (const float*)d_in[7];

  setup_kernel<<<1, 128, 0, stream>>>(W_enc, b_enc, W_rel, b_rel, W_root, W_dec, b_dec, d_ws);
  fused_mlp_kernel<<<GRID, BLOCK, 0, stream>>>(x, d_ws, (float*)d_out);
}
